// TokenformerAttention_12532714569897
// MI455X (gfx1250) — matrix operand order repaired and checked by reading, not compiled, not run on hardware
//
#include <hip/hip_runtime.h>
#include <hip/hip_bf16.h>
#include <math.h>

typedef __attribute__((ext_vector_type(16))) _Float16 v16h;
typedef __attribute__((ext_vector_type(8)))  _Float16 v8h;
typedef __attribute__((ext_vector_type(8)))  float    v8f;

#define DIM    1024
#define HEADS  16
#define DH     64
#define INNER  1024
#define NPT    1024
#define BATCH  2
#define SEQ    2048
#define ROWS   (BATCH * SEQ)   // 4096

// ---------------------------------------------------------------------------
// CDNA5 async global->LDS copy (16B per lane), tracked by ASYNCcnt.
// dst: wave-relative LDS byte offset (low 32 bits of a generic shared pointer),
// src: 64-bit global address. offset: applies to BOTH lds and mem addresses.
// ---------------------------------------------------------------------------
__device__ __forceinline__ void async_copy_b128(unsigned int lds_off,
                                                const void* gptr) {
  unsigned long long g = (unsigned long long)gptr;
  asm volatile("global_load_async_to_lds_b128 %0, %1, off"
               :: "v"(lds_off), "v"(g) : "memory");
}
__device__ __forceinline__ void async_copy_b128_o16(unsigned int lds_off,
                                                    const void* gptr) {
  unsigned long long g = (unsigned long long)gptr;
  asm volatile("global_load_async_to_lds_b128 %0, %1, off offset:16"
               :: "v"(lds_off), "v"(g) : "memory");
}
__device__ __forceinline__ void wait_async0() {
  asm volatile("s_wait_asynccnt 0x0" ::: "memory");
}

// ---------------------------------------------------------------------------
// WMMA fragment loaders (f16, 16x16x32, wave32).
// A layout (16x32): lane L (L<16) row M=L, K in {0..15}; lane L+16 same
// row, K in {16..31} (K-contiguous, matching B).
// B layout (32x16): lane L holds column N=L%16, K chunk = (L<16 ? 0..15 : 16..31),
// K-contiguous -> one 32B load per lane (NT GEMM: B operand = rows of weight).
// C/D layout: VGPR r, lane L: row = r + 8*(L>>4), col = L&15.
// ---------------------------------------------------------------------------
__device__ __forceinline__ v16h load_a_frag(const _Float16* src, int ld,
                                            int mBase, int kBase, int lane) {
  const int m    = mBase + (lane & 15);
  const int koff = kBase + ((lane >> 4) << 4);            // 0 or 16
  v8h x0 = *(const v8h*)(src + (size_t)m * ld + koff);
  v8h x1 = *(const v8h*)(src + (size_t)m * ld + koff + 8);
  v16h a;
#pragma unroll
  for (int i = 0; i < 8; ++i) { a[i] = x0[i]; a[i + 8] = x1[i]; }
  return a;
}

// ---------------------------------------------------------------------------
// NT GEMM: C[M,N] = scale * A[M,K] * B[N,K]^T   (A,B f16 row-major)
// mode 0: C f32 row-major.  mode 1: C f16 scattered to [B,H,SEQ,DH].
// mode 2: C f16 row-major.
// Block: 256 threads = 8 waves; tile 128x64; K-step 32.
// B tile staged with double-buffered async global->LDS DMA; A software-pipelined.
// ---------------------------------------------------------------------------
__global__ __launch_bounds__(256)
void gemm_nt_f16(const _Float16* __restrict__ A, const _Float16* __restrict__ B,
                 void* __restrict__ Cout, int M, int N, int K,
                 float scale, int mode)
{
  __shared__ _Float16 Bs[2][64 * 32];
  const int t     = threadIdx.x;
  const int lane  = t & 31;
  const int wave  = t >> 5;
  const int mBase = blockIdx.x * 128 + wave * 16;
  const int nBase = blockIdx.y * 64;

  // async B staging: thread t -> row br (0..63), 8 halfs (16B) at column bc
  const int br = t >> 2;
  const int bc = (t & 3) * 8;
  const _Float16* bRow = B + (size_t)(nBase + br) * K + bc;
  const unsigned int ldsB[2] = {
    (unsigned int)(size_t)&Bs[0][br * 32 + bc],
    (unsigned int)(size_t)&Bs[1][br * 32 + bc]
  };

  v8f zero = {};
  v8f acc[4] = {zero, zero, zero, zero};

  const int nSteps = K >> 5;
  async_copy_b128(ldsB[0], bRow);                 // prologue: tile 0 in flight
  v16h aF = load_a_frag(A, K, mBase, 0, lane);

  for (int i = 0; i < nSteps; ++i) {
    wait_async0();           // my async writes for tile i have landed
    __syncthreads();         // everyone's writes landed; prev tile's reads done
    if (i + 1 < nSteps)
      async_copy_b128(ldsB[(i + 1) & 1], bRow + ((i + 1) << 5));
    v16h aNext = aF;
    if (i + 1 < nSteps)
      aNext = load_a_frag(A, K, mBase, (i + 1) << 5, lane);

    const _Float16* BsI = &Bs[i & 1][0];
#pragma unroll
    for (int nt = 0; nt < 4; ++nt) {
      const int n    = nt * 16 + (lane & 15);
      const int koff = (lane >> 4) << 4;
      v16h bF = *(const v16h*)(&BsI[n * 32 + koff]);
      acc[nt] = __builtin_amdgcn_wmma_f32_16x16x32_f16(
          false, aF, false, bF, (short)0, acc[nt], false, false);
    }
    aF = aNext;
  }

#pragma unroll
  for (int nt = 0; nt < 4; ++nt) {
#pragma unroll
    for (int r = 0; r < 8; ++r) {
      const float v = acc[nt][r] * scale;
      const int m = mBase + r + ((lane >> 4) << 3);
      const int n = nBase + nt * 16 + (lane & 15);
      if (mode == 0) {
        ((float*)Cout)[(size_t)m * N + n] = v;
      } else if (mode == 1) {   // scatter to [B,H,SEQ,DH] f16
        const int b = m >> 11, tok = m & 2047;
        const int h = n >> 6,  d   = n & 63;
        ((_Float16*)Cout)[(((size_t)(b * HEADS + h)) * SEQ + tok) * DH + d] = (_Float16)v;
      } else {
        ((_Float16*)Cout)[(size_t)m * N + n] = (_Float16)v;
      }
    }
  }
}

// ---------------------------------------------------------------------------
// LayerNorm over DIM, output f16. One block per row.
// ---------------------------------------------------------------------------
__global__ __launch_bounds__(256)
void layernorm_f16(const float* __restrict__ x, const float* __restrict__ g,
                   const float* __restrict__ bt, _Float16* __restrict__ y)
{
  __shared__ float red[256];
  const int row = blockIdx.x;
  const float* xr = x + (size_t)row * DIM;
  float s = 0.f;
  for (int i = threadIdx.x; i < DIM; i += 256) s += xr[i];
  red[threadIdx.x] = s; __syncthreads();
  for (int o = 128; o > 0; o >>= 1) {
    if (threadIdx.x < o) red[threadIdx.x] += red[threadIdx.x + o];
    __syncthreads();
  }
  const float mu = red[0] * (1.0f / DIM);
  __syncthreads();
  float v = 0.f;
  for (int i = threadIdx.x; i < DIM; i += 256) { float d = xr[i] - mu; v += d * d; }
  red[threadIdx.x] = v; __syncthreads();
  for (int o = 128; o > 0; o >>= 1) {
    if (threadIdx.x < o) red[threadIdx.x] += red[threadIdx.x + o];
    __syncthreads();
  }
  const float rinv = rsqrtf(red[0] * (1.0f / DIM) + 1e-5f);
  for (int i = threadIdx.x; i < DIM; i += 256) {
    float o = (xr[i] - mu) * rinv * g[i] + bt[i];
    y[(size_t)row * DIM + i] = (_Float16)o;
  }
}

// ---------------------------------------------------------------------------
// Row L2-normalize (clamped at 1e-12) + exact GELU, output f16. Block per row.
// ---------------------------------------------------------------------------
__global__ __launch_bounds__(256)
void rownorm_gelu(const float* __restrict__ sim, _Float16* __restrict__ out)
{
  __shared__ float red[256];
  const int row = blockIdx.x;
  const float* sr = sim + (size_t)row * NPT;
  float s = 0.f;
  for (int i = threadIdx.x; i < NPT; i += 256) { float v = sr[i]; s += v * v; }
  red[threadIdx.x] = s; __syncthreads();
  for (int o = 128; o > 0; o >>= 1) {
    if (threadIdx.x < o) red[threadIdx.x] += red[threadIdx.x + o];
    __syncthreads();
  }
  const float rn = 1.0f / fmaxf(sqrtf(red[0]), 1e-12f);
  for (int i = threadIdx.x; i < NPT; i += 256) {
    float v  = sr[i] * rn;
    float gl = 0.5f * v * (1.0f + erff(v * 0.70710678118654752f));
    out[(size_t)row * NPT + i] = (_Float16)gl;
  }
}

// ---------------------------------------------------------------------------
// f32 -> f16 elementwise, and tiled f32 -> f16 transpose (1024x1024).
// ---------------------------------------------------------------------------
__global__ void conv_f16(const float* __restrict__ x, _Float16* __restrict__ y, int n) {
  int i = blockIdx.x * blockDim.x + threadIdx.x;
  if (i < n) y[i] = (_Float16)x[i];
}

__global__ __launch_bounds__(256)
void conv_f16_T(const float* __restrict__ x, _Float16* __restrict__ y)
{
  __shared__ _Float16 tile[32][33];
  const int bx = blockIdx.x * 32, by = blockIdx.y * 32;
  const int tx = threadIdx.x & 31, ty = threadIdx.x >> 5;   // 8 rows / pass
#pragma unroll
  for (int i = 0; i < 32; i += 8)
    tile[ty + i][tx] = (_Float16)x[(size_t)(by + ty + i) * INNER + bx + tx];
  __syncthreads();
#pragma unroll
  for (int i = 0; i < 32; i += 8)
    y[(size_t)(bx + ty + i) * NPT + by + tx] = tile[tx][ty + i];
}

// ---------------------------------------------------------------------------
// Flash attention. Q,K,V: [B*H, SEQ, DH] f16. out: [B*SEQ, INNER] f16.
// grid = (B*H, SEQ/64), block = 128 (4 waves). Wave owns a 16-query tile and
// iterates KV in tiles of 32 with online softmax. K tile staged via async DMA;
// V staged transposed [d][j] in LDS (manual scatter) for the PV B operand.
// ---------------------------------------------------------------------------
__global__ __launch_bounds__(128)
void flash_attn(const _Float16* __restrict__ Q, const _Float16* __restrict__ K,
                const _Float16* __restrict__ V, _Float16* __restrict__ out)
{
  __shared__ _Float16 Ks[32 * 64];      // K tile, row-major
  __shared__ _Float16 Vt[64 * 32];      // V tile, transposed [d][j]
  __shared__ _Float16 Ps[4][16 * 32];   // per-wave P scratch (C->A relayout)

  const int t    = threadIdx.x;
  const int lane = t & 31;
  const int wave = t >> 5;
  const int bh   = blockIdx.x;               // 0..31
  const int b    = bh >> 4, h = bh & 15;
  const int qBase = blockIdx.y * 64 + wave * 16;

  const _Float16* Qb = Q + (size_t)bh * SEQ * DH;
  const _Float16* Kb = K + (size_t)bh * SEQ * DH;
  const _Float16* Vb = V + (size_t)bh * SEQ * DH;

  const v16h aQ0 = load_a_frag(Qb, DH, qBase, 0, lane);
  const v16h aQ1 = load_a_frag(Qb, DH, qBase, 32, lane);

  v8f zero = {};
  v8f accO[4] = {zero, zero, zero, zero};
  float mrow[8], lrow[8];
#pragma unroll
  for (int r = 0; r < 8; ++r) { mrow[r] = -1e30f; lrow[r] = 0.f; }

  const float sm_scale = 0.125f;   // DH^-0.5
  const int   kr = t >> 2;         // staging row 0..31
  const int   kc = (t & 3) * 16;   // staging column (halfs), 32B per thread
  const unsigned int ldsK = (unsigned int)(size_t)&Ks[kr * 64 + kc];

  for (int j0 = 0; j0 < SEQ; j0 += 32) {
    __syncthreads();               // prev iteration's LDS reads all complete
    {
      const _Float16* kSrc = Kb + (size_t)(j0 + kr) * DH + kc;
      async_copy_b128(ldsK, kSrc);       // 16B
      async_copy_b128_o16(ldsK, kSrc);   // +16B (offset applies to lds & mem)
      // V tile transposed into Vt (needs [d][j] layout -> manual scatter)
      v16h vv = *(const v16h*)(Vb + (size_t)(j0 + kr) * DH + kc);
#pragma unroll
      for (int i = 0; i < 16; ++i) Vt[(kc + i) * 32 + kr] = vv[i];
    }
    wait_async0();
    __syncthreads();

    // S = Q * Ktile^T : two 16x16 score tiles, K-dim 64 = 2 wmma steps
    v8f s0 = zero, s1 = zero;
    {
      const int nloc  = lane & 15;
      const int khalf = (lane >> 4) << 4;
      v16h b00 = *(const v16h*)(&Ks[nloc * 64 + khalf]);
      v16h b01 = *(const v16h*)(&Ks[nloc * 64 + 32 + khalf]);
      v16h b10 = *(const v16h*)(&Ks[(16 + nloc) * 64 + khalf]);
      v16h b11 = *(const v16h*)(&Ks[(16 + nloc) * 64 + 32 + khalf]);
      s0 = __builtin_amdgcn_wmma_f32_16x16x32_f16(false, aQ0, false, b00, (short)0, s0, false, false);
      s0 = __builtin_amdgcn_wmma_f32_16x16x32_f16(false, aQ1, false, b01, (short)0, s0, false, false);
      s1 = __builtin_amdgcn_wmma_f32_16x16x32_f16(false, aQ0, false, b10, (short)0, s1, false, false);
      s1 = __builtin_amdgcn_wmma_f32_16x16x32_f16(false, aQ1, false, b11, (short)0, s1, false, false);
    }

    // online softmax; row mapping (r + 8*(lane>>4)) identical for S and O accs
#pragma unroll
    for (int r = 0; r < 8; ++r) {
      float sv0 = s0[r] * sm_scale, sv1 = s1[r] * sm_scale;
      float mx = fmaxf(sv0, sv1);
#pragma unroll
      for (int o = 8; o > 0; o >>= 1) mx = fmaxf(mx, __shfl_xor(mx, o, 16));
      const float mnew  = fmaxf(mrow[r], mx);
      const float alpha = __expf(mrow[r] - mnew);
      const float p0 = __expf(sv0 - mnew), p1 = __expf(sv1 - mnew);
      float ls = p0 + p1;
#pragma unroll
      for (int o = 8; o > 0; o >>= 1) ls += __shfl_xor(ls, o, 16);
      lrow[r] = alpha * lrow[r] + ls;
      mrow[r] = mnew;
      s0[r] = p0; s1[r] = p1;
#pragma unroll
      for (int dt = 0; dt < 4; ++dt) accO[dt][r] *= alpha;
    }

    {   // spill P (C layout) to per-wave LDS scratch
      _Float16* Pw = &Ps[wave][0];
      const int col  = lane & 15;
      const int rofs = (lane >> 4) << 3;
#pragma unroll
      for (int r = 0; r < 8; ++r) {
        Pw[(r + rofs) * 32 + col]      = (_Float16)s0[r];
        Pw[(r + rofs) * 32 + 16 + col] = (_Float16)s1[r];
      }
    }
    __syncthreads();
    {   // O += P * Vtile  (B operand from transposed V in LDS, K-contiguous)
      const v16h aP = load_a_frag(&Ps[wave][0], 32, 0, 0, lane);
      const int dloc  = lane & 15;
      const int jhalf = (lane >> 4) << 4;
#pragma unroll
      for (int dt = 0; dt < 4; ++dt) {
        v16h bV = *(const v16h*)(&Vt[(dt * 16 + dloc) * 32 + jhalf]);
        accO[dt] = __builtin_amdgcn_wmma_f32_16x16x32_f16(
            false, aP, false, bV, (short)0, accO[dt], false, false);
      }
    }
  }

#pragma unroll
  for (int dt = 0; dt < 4; ++dt) {
#pragma unroll
    for (int r = 0; r < 8; ++r) {
      const int qrow = qBase + r + ((lane >> 4) << 3);
      const int d    = dt * 16 + (lane & 15);
      const float o  = accO[dt][r] / lrow[r];
      out[((size_t)(b * SEQ + qrow)) * INNER + h * DH + d] = (_Float16)o;
    }
  }
}

// ---------------------------------------------------------------------------
extern "C" void kernel_launch(void* const* d_in, const int* in_sizes, int n_in,
                              void* d_out, int out_size, void* d_ws, size_t ws_size,
                              hipStream_t stream)
{
  const float* x     = (const float*)d_in[0];
  const float* ln_g  = (const float*)d_in[1];
  const float* ln_b  = (const float*)d_in[2];
  const float* q_key = (const float*)d_in[3];
  const float* q_val = (const float*)d_in[4];
  const float* k_key = (const float*)d_in[5];
  const float* k_val = (const float*)d_in[6];
  const float* v_key = (const float*)d_in[7];
  const float* v_val = (const float*)d_in[8];
  const float* o_key = (const float*)d_in[9];
  const float* o_val = (const float*)d_in[10];
  float* out = (float*)d_out;

  char* ws = (char*)d_ws;
  size_t off = 0;
  auto alloc = [&](size_t bytes) -> void* {
    void* p = ws + off;
    off += (bytes + 255) & ~(size_t)255;
    return p;
  };
  _Float16* xh   = (_Float16*)alloc((size_t)ROWS * DIM * 2);     // 8 MB
  _Float16* wkh  = (_Float16*)alloc((size_t)NPT * DIM * 2);      // 2 MB (reused)
  _Float16* wvTh = (_Float16*)alloc((size_t)INNER * NPT * 2);    // 2 MB (reused)
  float*    sim  = (float*)   alloc((size_t)ROWS * NPT * 4);     // 16 MB (reused)
  _Float16* atth = (_Float16*)alloc((size_t)ROWS * NPT * 2);     // 8 MB (reused)
  _Float16* qh   = (_Float16*)alloc((size_t)ROWS * INNER * 2);   // 8 MB
  _Float16* kh   = (_Float16*)alloc((size_t)ROWS * INNER * 2);   // 8 MB
  _Float16* vh   = (_Float16*)alloc((size_t)ROWS * INNER * 2);   // 8 MB
  _Float16* outh = (_Float16*)alloc((size_t)ROWS * INNER * 2);   // 8 MB

  const dim3 gGemm(ROWS / 128, 1024 / 64);   // (32, 16)
  const float kscale = 0.03125f;             // 1024^-0.5

  layernorm_f16<<<ROWS, 256, 0, stream>>>(x, ln_g, ln_b, xh);

  struct Stage { const float* key; const float* val; _Float16* dst; };
  Stage st[3] = { {q_key, q_val, qh}, {k_key, k_val, kh}, {v_key, v_val, vh} };
  for (int s = 0; s < 3; ++s) {
    conv_f16<<<(NPT * DIM + 255) / 256, 256, 0, stream>>>(st[s].key, wkh, NPT * DIM);
    conv_f16_T<<<dim3(32, 32), 256, 0, stream>>>(st[s].val, wvTh);
    gemm_nt_f16<<<gGemm, 256, 0, stream>>>(xh, wkh, sim, ROWS, NPT, DIM, kscale, 0);
    rownorm_gelu<<<ROWS, 256, 0, stream>>>(sim, atth);
    gemm_nt_f16<<<gGemm, 256, 0, stream>>>(atth, wvTh, st[s].dst, ROWS, INNER, NPT, 1.0f, 1);
  }

  flash_attn<<<dim3(BATCH * HEADS, SEQ / 64), 128, 0, stream>>>(qh, kh, vh, outh);

  conv_f16<<<(NPT * INNER + 255) / 256, 256, 0, stream>>>(o_key, wkh, NPT * INNER);
  conv_f16_T<<<dim3(32, 32), 256, 0, stream>>>(o_val, wvTh);
  gemm_nt_f16<<<gGemm, 256, 0, stream>>>(outh, wkh, sim, ROWS, NPT, INNER, kscale, 0);
  rownorm_gelu<<<ROWS, 256, 0, stream>>>(sim, atth);
  gemm_nt_f16<<<gGemm, 256, 0, stream>>>(atth, wvTh, out, ROWS, DIM, NPT, 1.0f, 0);
}
